// FastKANLayer_46385646796842
// MI455X (gfx1250) — matrix operand order repaired and checked by reading, not compiled, not run on hardware
//
#include <hip/hip_runtime.h>

// FastKAN layer as a single fused f16 WMMA GEMM on gfx1250 (wave32).
// out[4096,512] = G[4096, 8704] x Bpack[8704, 512]
//   K 0..8191   : spline basis (i*16+n), coeffs*ws folded, extrapolation folded into G
//   K 8192..8703: silu(x) bypass against wb
// B tiles are staged to LDS with GLOBAL_LOAD_ASYNC_TO_LDS_B128 (ASYNCcnt),
// A tiles are built in LDS by cooperative de Boor evaluation; WMMAs read both from LDS.
typedef _Float16 f16;
typedef __attribute__((ext_vector_type(16))) _Float16 v16h;
typedef __attribute__((ext_vector_type(8)))  _Float16 v8h;
typedef __attribute__((ext_vector_type(8)))  float    v8f;

#define DEV __device__ __forceinline__

// ---------------- compile-time boundary basis (replicates reference recursion) --------------
struct B16d { double v[16]; };
constexpr double knotd(int i) { return (i <= 3) ? 0.0 : ((i >= 16) ? 1.0 : (double)(i - 3) / 13.0); }
constexpr B16d basis_cx(double x) {
  double B[16] = {};
  for (int n = 0; n < 16; n++) B[n] = (x >= knotd(n) && x < knotd(n + 1)) ? 1.0 : 0.0;
  B[15] = (x >= knotd(15) && x <= knotd(16)) ? 1.0 : 0.0;
  for (int p = 1; p <= 3; p++) {
    double Bn[16] = {};
    for (int n = 0; n < 16; n++) {
      double d1 = knotd(n + p) - knotd(n);
      double d2 = knotd(n + p + 1) - knotd(n + 1);
      double a = (d1 > 1e-12 || d1 < -1e-12) ? (x - knotd(n)) / d1 : 0.0;
      double c = (d2 > 1e-12 || d2 < -1e-12) ? (knotd(n + p + 1) - x) / d2 : 0.0;
      Bn[n] = a * B[n] + c * ((n < 15) ? B[n + 1] : 0.0);
    }
    for (int n = 0; n < 16; n++) B[n] = Bn[n];
  }
  B16d r{}; for (int n = 0; n < 16; n++) r.v[n] = B[n]; return r;
}
constexpr double EPS_D = 0.001;
constexpr B16d BMIN  = basis_cx(0.0);        // == e0
constexpr B16d BMAX  = basis_cx(1.0);        // == e15
constexpr B16d BMINE = basis_cx(EPS_D);
constexpr B16d BMAXE = basis_cx(1.0 - EPS_D);
constexpr float SLO(int d) { return (float)((BMINE.v[d] - BMIN.v[d]) / (EPS_D + 1e-12)); }
constexpr float SHI(int d) { return (float)((BMAX.v[12 + d] - BMAXE.v[12 + d]) / (EPS_D + 1e-12)); }
constexpr float BLO(int d) { return (float)BMIN.v[d]; }
constexpr float BHI(int d) { return (float)BMAX.v[12 + d]; }

// ---------------- device basis: 4 nonzero values via triangular de Boor --------------------
DEV float knotf(int i) { int m = i - 3; m = m < 0 ? 0 : m; m = m > 13 ? 13 : m; return (float)m * (1.0f / 13.0f); }

DEV void eval4(float x, int& j, float v[4]) {
  if (x < 0.0f) {                    // linear extrapolation below
    j = 3;
    v[0] = BLO(0) + x * SLO(0); v[1] = BLO(1) + x * SLO(1);
    v[2] = BLO(2) + x * SLO(2); v[3] = BLO(3) + x * SLO(3);
  } else if (x > 1.0f) {             // linear extrapolation above
    j = 15; float d = x - 1.0f;
    v[0] = BHI(0) + d * SHI(0); v[1] = BHI(1) + d * SHI(1);
    v[2] = BHI(2) + d * SHI(2); v[3] = BHI(3) + d * SHI(3);
  } else {                           // de Boor (NURBS book A2.2), span j, nonzeros n = j-3..j
    int m = (int)(x * 13.0f); m = m > 12 ? 12 : m; j = 3 + m;
    float N0 = 1.0f, N1, N2, N3, t, saved;
    const float l1 = x - knotf(j),     r1 = knotf(j + 1) - x;
    const float l2 = x - knotf(j - 1), r2 = knotf(j + 2) - x;
    const float l3 = x - knotf(j - 2), r3 = knotf(j + 3) - x;
    t = N0 / (r1 + l1); N0 = r1 * t; N1 = l1 * t;                        // p=1
    t = N0 / (r1 + l2); N0 = r1 * t; saved = l2 * t;                     // p=2
    t = N1 / (r2 + l1); N1 = saved + r2 * t; N2 = l1 * t;
    t = N0 / (r1 + l3); N0 = r1 * t; saved = l3 * t;                     // p=3
    t = N1 / (r2 + l2); N1 = saved + r2 * t; saved = l2 * t;
    t = N2 / (r3 + l1); N2 = saved + r3 * t; N3 = l1 * t;
    v[0] = N0; v[1] = N1; v[2] = N2; v[3] = N3;
  }
}

// ---------------- kernel 1: pack B = [coeffs*ws ; wb] as f16 in WMMA B-fragment layout -----
// frag f = ks*32 + (o/16), 512 halves: lane (o%16)+((kk/8)%2)*16, half = (kk/16)*8 + kk%8, kk = K%32
__global__ __launch_bounds__(256) void kan_pack_kernel(
    const float* __restrict__ coeffs, const float* __restrict__ ws,
    const float* __restrict__ wb, f16* __restrict__ bpack)
{
  const int idx = blockIdx.x * 256 + threadIdx.x;   // 0 .. 8704*512-1
  const int k = idx >> 9;
  const int o = idx & 511;
  float v;
  if (k < 8192) {
    const int i = k >> 4, n = k & 15;
    v = coeffs[((size_t)(i * 512 + o) << 4) + n] * ws[i * 512 + o];
  } else {
    v = wb[(k - 8192) * 512 + o];
  }
  const int ks = k >> 5, kk = k & 31;
  const int lane = (o & 15) + (((kk >> 3) & 1) << 4);
  bpack[((size_t)(ks * 32 + (o >> 4)) << 9) + lane * 16 + ((kk >> 4) << 3) + (kk & 7)] = (f16)v;
}

// ---------------- kernel 2: fused basis-eval + async-staged WMMA GEMM ----------------------
#define PH_BASIS 64
#define PH_TOTAL 68
#define FRAG_B   1024                       // one 16x32 f16 fragment = 32 lanes * 32B
#define APHASE_B (16 * FRAG_B)              // A: 4 m-subtiles * 4 K-steps = 16KB
#define BPHASE_B (32 * FRAG_B)              // B: 4 K-steps * 8 n-subtiles = 32KB

#define WMMA16(AF, BF, CF) __builtin_amdgcn_wmma_f32_16x16x32_f16( \
    false, (AF), false, (BF), (short)0, (CF), false, false)

__global__ __launch_bounds__(128, 1) void kan_wmma_kernel(
    const float* __restrict__ x, const f16* __restrict__ bpack, float* __restrict__ out)
{
  __shared__ alignas(64) char asm_buf[2][APHASE_B];   // A fragments (basis / silu)
  __shared__ alignas(64) char bsm_buf[2][BPHASE_B];   // B fragments (async-staged)
  const int tid   = threadIdx.x;
  const int lane  = tid & 31;
  const int w     = tid >> 5;               // wave id 0..3
  const int bm    = blockIdx.x & 63;        // 64 M-tiles of 64 rows
  const int bn    = blockIdx.x >> 6;        // 4 N-tiles of 128 cols
  const int row0  = bm * 64;
  const int b_loc = tid >> 1;               // staging row 0..63
  const int mtile = b_loc >> 4;
  const int lrow  = b_loc & 15;
  const size_t xrow = (size_t)(row0 + b_loc) * 512;

  // wave tile: 32 (2 m-subtiles) x 64 (4 n-subtiles) -> B frags reused 2x, A frags 4x
  const int wm = (w & 1) * 2;               // first m-subtile of this wave
  const int wn = (w >> 1) * 4;              // first n-subtile of this wave

  v8f acc[8] = {};                          // acc[mi*4+on]

  // stage 8 input columns (i = p*8 .. p*8+7) of spline basis into A-fragment layout
  auto stage_basis = [&](int p, char* buf) {
    const float4 xv = *(const float4*)(x + xrow + p * 8 + (tid & 1) * 4);
    const float xs[4] = {xv.x, xv.y, xv.z, xv.w};
#pragma unroll
    for (int q = 0; q < 4; q++) {
      const int il = (tid & 1) * 4 + q;     // i within phase, 0..7
      const int s  = il >> 1;               // K-step 0..3
      char* low = buf + (mtile * 4 + s) * FRAG_B + lrow * 32 + (il & 1) * 16;
      *(uint4*)low         = make_uint4(0u, 0u, 0u, 0u);   // n = 0..7  slot
      *(uint4*)(low + 512) = make_uint4(0u, 0u, 0u, 0u);   // n = 8..15 slot (lane+16)
      int j; float v[4];
      eval4(xs[q], j, v);
#pragma unroll
      for (int d = 0; d < 4; d++) {
        const int n = j - 3 + d;
        *(f16*)(low + ((n & 7) * 2) + ((n >> 3) * 512)) = (f16)v[d];
      }
    }
  };

  // stage 128 silu(x) columns (K = 8192.. region) into A-fragment layout
  auto stage_silu = [&](int sp, char* buf) {
#pragma unroll
    for (int q = 0; q < 8; q++) {
      const int kc = (tid & 1) * 8 + q;     // 16B chunk 0..15 along K
      const int cc = kc * 8;
      const float4 a = *(const float4*)(x + xrow + sp * 128 + cc);
      const float4 b = *(const float4*)(x + xrow + sp * 128 + cc + 4);
      const float f[8] = {a.x, a.y, a.z, a.w, b.x, b.y, b.z, b.w};
      v8h h;
#pragma unroll
      for (int e = 0; e < 8; e++) { const float xe = f[e]; h[e] = (f16)(xe / (1.0f + __expf(-xe))); }
      const int s = kc >> 2, sub = kc & 3;
      char* dst = buf + (mtile * 4 + s) * FRAG_B + (lrow + (sub & 1) * 16) * 32 + (sub >> 1) * 16;
      *(v8h*)dst = h;
    }
  };

  // async-stage one phase of B (32KB) into LDS: wave w copies K-step row s=w (8KB).
  // Fragments are contiguous in bpack, and the instruction offset applies to BOTH the
  // global and the LDS address, so one address pair per wave covers all 16 chunks.
  auto stage_b_async = [&](int pn, int bufsel) {
    const char* src = (const char*)bpack
                    + ((size_t)((pn * 4 + w) * 32 + bn * 8) << 10) + lane * 16;
    const unsigned dst = (unsigned)(uintptr_t)(&bsm_buf[bufsel][0])
                       + (unsigned)(w * 8192 + lane * 16);
    asm volatile("global_load_async_to_lds_b128 %0, %1, off offset:0\n\t"
                 "global_load_async_to_lds_b128 %0, %1, off offset:512\n\t"
                 "global_load_async_to_lds_b128 %0, %1, off offset:1024\n\t"
                 "global_load_async_to_lds_b128 %0, %1, off offset:1536\n\t"
                 "global_load_async_to_lds_b128 %0, %1, off offset:2048\n\t"
                 "global_load_async_to_lds_b128 %0, %1, off offset:2560\n\t"
                 "global_load_async_to_lds_b128 %0, %1, off offset:3072\n\t"
                 "global_load_async_to_lds_b128 %0, %1, off offset:3584\n\t"
                 "global_load_async_to_lds_b128 %0, %1, off offset:4096\n\t"
                 "global_load_async_to_lds_b128 %0, %1, off offset:4608\n\t"
                 "global_load_async_to_lds_b128 %0, %1, off offset:5120\n\t"
                 "global_load_async_to_lds_b128 %0, %1, off offset:5632\n\t"
                 "global_load_async_to_lds_b128 %0, %1, off offset:6144\n\t"
                 "global_load_async_to_lds_b128 %0, %1, off offset:6656\n\t"
                 "global_load_async_to_lds_b128 %0, %1, off offset:7168\n\t"
                 "global_load_async_to_lds_b128 %0, %1, off offset:7680"
                 :: "v"(dst), "v"(src) : "memory");
  };

  // consume one phase: 4 K-steps; per K-step load 2 A + 4 B fragments into distinct
  // registers (keep-alive asm pins them), then 8 back-to-back WMMAs reuse them.
  auto compute = [&](int p) {
    const char* abuf = asm_buf[p & 1] + lane * 32;
    const char* bbuf = bsm_buf[p & 1] + lane * 32;
#pragma unroll
    for (int s = 0; s < 4; s++) {
      v16h A0 = *(const v16h*)(abuf + ((wm + 0) * 4 + s) * FRAG_B);
      v16h A1 = *(const v16h*)(abuf + ((wm + 1) * 4 + s) * FRAG_B);
      v16h B0 = *(const v16h*)(bbuf + (s * 8 + wn + 0) * FRAG_B);
      v16h B1 = *(const v16h*)(bbuf + (s * 8 + wn + 1) * FRAG_B);
      v16h B2 = *(const v16h*)(bbuf + (s * 8 + wn + 2) * FRAG_B);
      v16h B3 = *(const v16h*)(bbuf + (s * 8 + wn + 3) * FRAG_B);
      asm volatile("" :: "v"(A0), "v"(A1), "v"(B0), "v"(B1), "v"(B2), "v"(B3));
      acc[0] = WMMA16(A0, B0, acc[0]);
      acc[1] = WMMA16(A0, B1, acc[1]);
      acc[2] = WMMA16(A0, B2, acc[2]);
      acc[3] = WMMA16(A0, B3, acc[3]);
      acc[4] = WMMA16(A1, B0, acc[4]);
      acc[5] = WMMA16(A1, B1, acc[5]);
      acc[6] = WMMA16(A1, B2, acc[6]);
      acc[7] = WMMA16(A1, B3, acc[7]);
    }
  };

  stage_basis(0, asm_buf[0]);
  stage_b_async(0, 0);
  for (int p = 0; p < PH_TOTAL; p++) {
    // all of this wave's async copies (staged for phase p last iteration) must land
    // before we signal the barrier that lets other waves read them.
    asm volatile("s_wait_asynccnt 0x0" ::: "memory");
    __syncthreads();
    const int pn = p + 1;
    if (pn < PH_BASIS)      stage_basis(pn, asm_buf[pn & 1]);
    else if (pn < PH_TOTAL) stage_silu(pn - PH_BASIS, asm_buf[pn & 1]);
    if (pn < PH_TOTAL)      stage_b_async(pn, pn & 1);
    compute(p);
  }

  // epilogue: C/D layout -> global (VGPR r: lanes0-15 M=r, lanes16-31 M=r+8)
  const int col0 = bn * 128 + (w >> 1) * 64 + (lane & 15);
  const int rb   = row0 + (w & 1) * 32 + ((lane >> 4) << 3);
#pragma unroll
  for (int mi = 0; mi < 2; mi++) {
#pragma unroll
    for (int on = 0; on < 4; on++) {
#pragma unroll
      for (int r = 0; r < 8; r++) {
        out[(size_t)(rb + mi * 16 + r) * 512 + col0 + on * 16] = acc[mi * 4 + on][r];
      }
    }
  }
}

extern "C" void kernel_launch(void* const* d_in, const int* in_sizes, int n_in,
                              void* d_out, int out_size, void* d_ws, size_t ws_size,
                              hipStream_t stream) {
  const float* x      = (const float*)d_in[0];   // [4096, 512]
  const float* coeffs = (const float*)d_in[1];   // [512, 512, 16]
  const float* ws     = (const float*)d_in[2];   // [512, 512]
  const float* wb     = (const float*)d_in[3];   // [512, 512]
  float* out = (float*)d_out;                    // [4096, 512]
  f16* bpack = (f16*)d_ws;                       // 8704*512 f16 = 8.9 MB

  kan_pack_kernel<<<(8704 * 512) / 256, 256, 0, stream>>>(coeffs, ws, wb, bpack);
  kan_wmma_kernel<<<256, 128, 0, stream>>>(x, bpack, out);
}
